// DeepInterestNetwork_74380243632888
// MI455X (gfx1250) — compile-verified, matrix operand order checked by reading, not run on hardware
//
#include <hip/hip_runtime.h>
#include <hip/hip_bf16.h>

#define NF 20
#define VOC 200001   // V + 1
#define D 64
#define BATCH 4096
#define L 19
#define DENSE 64
#define DNN_IN 1408  // DENSE + D*(NF+1)
#define MROWS (BATCH * L)   // 77824, multiple of 16

// Async global->LDS staging for the DNN layer-1 A slab (CDNA5 path).
// Set to 0 to fall back to plain VGPR staging if the mnemonic fails to assemble.
#define ASYNC_LDS 1

typedef _Float16 h8v  __attribute__((ext_vector_type(8)));
typedef _Float16 v16h __attribute__((ext_vector_type(16)));
typedef float    v8f  __attribute__((ext_vector_type(8)));

// ---------------------------------------------------------------------------
// Load a 16x32 f16 tile (row-major source, leading dim ld) into the CDNA5
// WMMA A/B register layout:
//   lanes 0-15 : row r = lane,     vector elems 0-7 = K 0..7,  8-15 = K 16..23
//   lanes 16-31: row r = lane-16,  vector elems 0-7 = K 8..15, 8-15 = K 24..31
// Both halves are contiguous 16-byte loads -> global_load_b128 / ds_load_b128.
// For the B matrix we load from a pre-transposed weight (wT[n][k]), so a
// "row" here is an output column and the same layout applies.
// ---------------------------------------------------------------------------
__device__ inline v16h load_tile16x32(const _Float16* __restrict__ base, int ld, int lane) {
    int r  = lane & 15;
    int ko = (lane >> 4) * 8;
    const _Float16* p = base + (size_t)r * ld + ko;
    h8v lo = *(const h8v*)(p);
    h8v hi = *(const h8v*)(p + 16);
    return __builtin_shufflevector(lo, hi, 0,1,2,3,4,5,6,7,8,9,10,11,12,13,14,15);
}

__device__ inline v8f wmma_f16(v16h a, v16h b, v8f c) {
    return __builtin_amdgcn_wmma_f32_16x16x32_f16(false, a, false, b, (short)0, c, false, false);
}

// ---------------------------------------------------------------------------
// Kernel 0: convert + transpose weights to f16 (tiny, one pass)
//   aw1 (128x64)->aw1T (64x128), aw2 (64x16)->aw2T (16x64),
//   dw1 (1408x128)->dw1T (128x1408), dw2 (128x64)->dw2T (64x128)
// ---------------------------------------------------------------------------
__global__ void convert_weights(const float* __restrict__ aw1, const float* __restrict__ aw2,
                                const float* __restrict__ dw1, const float* __restrict__ dw2,
                                _Float16* __restrict__ aw1T, _Float16* __restrict__ aw2T,
                                _Float16* __restrict__ dw1T, _Float16* __restrict__ dw2T) {
    int t = blockIdx.x * blockDim.x + threadIdx.x;
    if (t < 64 * 128) { int n = t / 128, k = t % 128; aw1T[t] = (_Float16)aw1[k * 64 + n]; return; }
    t -= 64 * 128;
    if (t < 16 * 64)  { int n = t / 64,  k = t % 64;  aw2T[t] = (_Float16)aw2[k * 16 + n]; return; }
    t -= 16 * 64;
    if (t < 128 * 1408) { int n = t / 1408, k = t % 1408; dw1T[t] = (_Float16)dw1[k * 128 + n]; return; }
    t -= 128 * 1408;
    if (t < 64 * 128) { int n = t / 128, k = t % 128; dw2T[t] = (_Float16)dw2[k * 64 + n]; return; }
}

// ---------------------------------------------------------------------------
// Kernel 1: embedding gathers (fp32 -> f16), build
//   x[b, 0:64]              = dense
//   x[b, 64+f*64 : ...]     = embeds[b,f]            (f = 0..19)
//   a_in[b*19+l, 0:64]      = query (= embeds[b,0])
//   a_in[b*19+l, 64:128]    = hist_emb[b,l]  (table l+1)
// One block per batch row, 64 threads = one embedding lane each.
// ---------------------------------------------------------------------------
__global__ __launch_bounds__(64) void gather_prep(
    const float* __restrict__ dense, const int* __restrict__ sparse,
    const int* __restrict__ history, const float* __restrict__ emb,
    _Float16* __restrict__ x, _Float16* __restrict__ a_in) {
    int b = blockIdx.x;
    int d = threadIdx.x;
    _Float16* xb = x + (size_t)b * DNN_IN;
    xb[d] = (_Float16)dense[(size_t)b * DENSE + d];
    _Float16 q = (_Float16)0.0f;
#pragma unroll
    for (int f = 0; f < NF; ++f) {
        int idx = sparse[b * NF + f];
        float v = emb[((size_t)f * VOC + idx) * D + d];
        _Float16 hv = (_Float16)v;
        xb[DENSE + f * D + d] = hv;
        if (f == 0) q = hv;
    }
#pragma unroll
    for (int l = 0; l < L; ++l) {
        int hid = history[b * L + l];
        float v = emb[((size_t)(l + 1) * VOC + hid) * D + d];
        _Float16* row = a_in + (size_t)(b * L + l) * 128;
        row[d] = q;
        row[64 + d] = (_Float16)v;
    }
}

// ---------------------------------------------------------------------------
// Kernel 2: attention layer 1: (77824 x 128) @ (128 x 64) + ab1, ReLU -> h1 f16
// One block = one 16-row M tile; 4 waves = 4 N tiles; K loop = 4 WMMAs.
// ---------------------------------------------------------------------------
__global__ __launch_bounds__(128) void att_layer1(
    const _Float16* __restrict__ a_in, const _Float16* __restrict__ aw1T,
    const float* __restrict__ ab1, _Float16* __restrict__ h1) {
    int wave = threadIdx.x >> 5, lane = threadIdx.x & 31;
    int m_base = blockIdx.x * 16;
    int n_base = wave * 16;
    v8f c = {};
#pragma unroll
    for (int k = 0; k < 128; k += 32) {
        v16h a = load_tile16x32(a_in + (size_t)m_base * 128 + k, 128, lane);
        v16h b = load_tile16x32(aw1T + (size_t)n_base * 128 + k, 128, lane);
        c = wmma_f16(a, b, c);
    }
    int col = lane & 15, rb = (lane >> 4) * 8;
    float bias = ab1[n_base + col];
#pragma unroll
    for (int r = 0; r < 8; ++r) {
        float v = c[r] + bias;
        v = v > 0.f ? v : 0.f;
        h1[(size_t)(m_base + rb + r) * 64 + n_base + col] = (_Float16)v;
    }
}

// ---------------------------------------------------------------------------
// Kernel 3: attention layer 2 (64->16, WMMA, ReLU) fused with layer 3 (16->1)
// via cross-lane shuffle dot with aw3. One wave = one 16-row M tile.
// ---------------------------------------------------------------------------
__global__ __launch_bounds__(256) void att_layer23(
    const _Float16* __restrict__ h1, const _Float16* __restrict__ aw2T,
    const float* __restrict__ ab2, const float* __restrict__ aw3,
    const float* __restrict__ ab3, float* __restrict__ scores) {
    int wave = threadIdx.x >> 5, lane = threadIdx.x & 31;
    int m_base = (blockIdx.x * 8 + wave) * 16;
    v8f c = {};
#pragma unroll
    for (int k = 0; k < 64; k += 32) {
        v16h a = load_tile16x32(h1 + (size_t)m_base * 64 + k, 64, lane);
        v16h b = load_tile16x32(aw2T + k, 64, lane);   // single 16-col N tile
        c = wmma_f16(a, b, c);
    }
    int col = lane & 15, rb = (lane >> 4) * 8;
    float bias = ab2[col];
    float w3 = aw3[col];
    float p[8];
#pragma unroll
    for (int r = 0; r < 8; ++r) {
        float v = c[r] + bias;
        v = v > 0.f ? v : 0.f;
        p[r] = v * w3;
    }
#pragma unroll
    for (int m = 1; m < 16; m <<= 1) {
#pragma unroll
        for (int r = 0; r < 8; ++r) p[r] += __shfl_xor(p[r], m, 32);
    }
    if ((lane & 15) == 0) {
        float b3 = ab3[0];
#pragma unroll
        for (int r = 0; r < 8; ++r) scores[m_base + rb + r] = p[r] + b3;
    }
}

// ---------------------------------------------------------------------------
// Kernel 4: masked softmax over L=19 + weighted pooling of hist_emb
// (stored as the second half of a_in rows); writes pooled into x tail.
// One wave32 per batch row; lane l owns score l, lanes own 2 dims each for pool.
// ---------------------------------------------------------------------------
__global__ __launch_bounds__(32) void softmax_pool(
    const float* __restrict__ scores, const int* __restrict__ hist_len,
    const _Float16* __restrict__ a_in, _Float16* __restrict__ x) {
    int b = blockIdx.x;
    int lane = threadIdx.x;
    int hl = hist_len[b];
    const float NEG = -__builtin_inff();
    float s = (lane < L && lane < hl) ? scores[b * L + lane] : NEG;
    float m = s;
#pragma unroll
    for (int o = 16; o > 0; o >>= 1) m = fmaxf(m, __shfl_xor(m, o, 32));
    float e = (s == NEG) ? 0.f : __expf(s - m);
    float sum = e;
#pragma unroll
    for (int o = 16; o > 0; o >>= 1) sum += __shfl_xor(sum, o, 32);
    float w = e / sum;
    float p0 = 0.f, p1 = 0.f;
#pragma unroll
    for (int l = 0; l < L; ++l) {
        float wl = __shfl(w, l, 32);
        const _Float16* he = a_in + (size_t)(b * L + l) * 128 + 64;
        p0 += wl * (float)he[lane];
        p1 += wl * (float)he[lane + 32];
    }
    _Float16* xp = x + (size_t)b * DNN_IN + DENSE + NF * D;
    xp[lane] = (_Float16)p0;
    xp[lane + 32] = (_Float16)p1;
}

// ---------------------------------------------------------------------------
// Kernel 5: DNN layer 1: (4096 x 1408) @ (1408 x 128) + db1, ReLU -> h1d f16
// One block = one 16-row M tile, 8 waves = 8 N tiles, 44 K steps.
// The 16x1408 A slab (45 KB) is staged into LDS once per block with
// global_load_async_to_lds_b128 (ASYNCcnt path) so the 8 waves don't each
// re-fetch it from global. LDS rows padded 1408->1416 halfs so the 16
// row-lanes of a ds_load_b128 tile fetch spread across all 64 LDS banks.
// ---------------------------------------------------------------------------
#define LDS_LD 1416

__global__ __launch_bounds__(256) void dnn_layer1(
    const _Float16* __restrict__ x, const _Float16* __restrict__ dw1T,
    const float* __restrict__ db1, _Float16* __restrict__ h1d) {
    __shared__ __align__(16) _Float16 As[16 * LDS_LD];
    int wave = threadIdx.x >> 5, lane = threadIdx.x & 31;
    int m_base = blockIdx.x * 16;
    int n_base = wave * 16;

    // Stage A: 16 rows x 1408 halfs = 2816 16-byte chunks; 176 chunks per row.
    {
        const char* gbase = (const char*)(x + (size_t)m_base * DNN_IN);
#pragma unroll
        for (int i = 0; i < 11; ++i) {
            int c = (int)threadIdx.x + i * 256;
            int row = c / 176, colc = c % 176;
            const char* g = gbase + ((size_t)row * DNN_IN + colc * 8) * 2;
#if ASYNC_LDS
            unsigned int loff =
                (unsigned int)(uintptr_t)(&As[row * LDS_LD + colc * 8]);
            asm volatile("global_load_async_to_lds_b128 %0, %1, off"
                         :: "v"(loff), "v"(g)
                         : "memory");
#else
            h8v tmp = *(const h8v*)g;
            *(h8v*)(&As[row * LDS_LD + colc * 8]) = tmp;
#endif
        }
#if ASYNC_LDS
        asm volatile("s_wait_asynccnt 0x0" ::: "memory");
#endif
        __syncthreads();
    }

    v8f c = {};
    for (int k = 0; k < DNN_IN; k += 32) {
        v16h a = load_tile16x32(&As[k], LDS_LD, lane);                       // ds_load_b128
        v16h b = load_tile16x32(dw1T + (size_t)n_base * DNN_IN + k, DNN_IN, lane);
        c = wmma_f16(a, b, c);
    }
    int col = lane & 15, rb = (lane >> 4) * 8;
    float bias = db1[n_base + col];
#pragma unroll
    for (int r = 0; r < 8; ++r) {
        float v = c[r] + bias;
        v = v > 0.f ? v : 0.f;
        h1d[(size_t)(m_base + rb + r) * 128 + n_base + col] = (_Float16)v;
    }
}

// ---------------------------------------------------------------------------
// Kernel 6: DNN layer 2 (128->64, WMMA, ReLU) fused with layer 3 (64->1):
// 4 waves cover the 4 N tiles of one 16-row M tile; each wave shuffle-reduces
// its 16-column partial dot with dw3, partials combined through LDS.
// ---------------------------------------------------------------------------
__global__ __launch_bounds__(128) void dnn_layer23(
    const _Float16* __restrict__ h1d, const _Float16* __restrict__ dw2T,
    const float* __restrict__ db2, const float* __restrict__ dw3,
    const float* __restrict__ db3, float* __restrict__ out) {
    __shared__ float partial[4][16];
    int wave = threadIdx.x >> 5, lane = threadIdx.x & 31;
    int m_base = blockIdx.x * 16;
    int n_base = wave * 16;
    v8f c = {};
#pragma unroll
    for (int k = 0; k < 128; k += 32) {
        v16h a = load_tile16x32(h1d + (size_t)m_base * 128 + k, 128, lane);
        v16h b = load_tile16x32(dw2T + (size_t)n_base * 128 + k, 128, lane);
        c = wmma_f16(a, b, c);
    }
    int col = lane & 15, rb = (lane >> 4) * 8;
    float bias = db2[n_base + col];
    float w3 = dw3[n_base + col];
    float p[8];
#pragma unroll
    for (int r = 0; r < 8; ++r) {
        float v = c[r] + bias;
        v = v > 0.f ? v : 0.f;
        p[r] = v * w3;
    }
#pragma unroll
    for (int m = 1; m < 16; m <<= 1) {
#pragma unroll
        for (int r = 0; r < 8; ++r) p[r] += __shfl_xor(p[r], m, 32);
    }
    if ((lane & 15) == 0) {
#pragma unroll
        for (int r = 0; r < 8; ++r) partial[wave][rb + r] = p[r];
    }
    __syncthreads();
    if (threadIdx.x < 16) {
        out[m_base + threadIdx.x] = partial[0][threadIdx.x] + partial[1][threadIdx.x] +
                                    partial[2][threadIdx.x] + partial[3][threadIdx.x] + db3[0];
    }
}

// ---------------------------------------------------------------------------
extern "C" void kernel_launch(void* const* d_in, const int* in_sizes, int n_in,
                              void* d_out, int out_size, void* d_ws, size_t ws_size,
                              hipStream_t stream) {
    const float* dense    = (const float*)d_in[0];
    const int*   sparse   = (const int*)  d_in[1];
    const int*   history  = (const int*)  d_in[2];
    const int*   hist_len = (const int*)  d_in[3];
    const float* emb      = (const float*)d_in[4];
    const float* aw1 = (const float*)d_in[5];
    const float* ab1 = (const float*)d_in[6];
    const float* aw2 = (const float*)d_in[7];
    const float* ab2 = (const float*)d_in[8];
    const float* aw3 = (const float*)d_in[9];
    const float* ab3 = (const float*)d_in[10];
    const float* dw1 = (const float*)d_in[11];
    const float* db1 = (const float*)d_in[12];
    const float* dw2 = (const float*)d_in[13];
    const float* db2 = (const float*)d_in[14];
    const float* dw3 = (const float*)d_in[15];
    const float* db3 = (const float*)d_in[16];
    float* out = (float*)d_out;

    char* ws = (char*)d_ws;
    size_t off = 0;
    auto take = [&](size_t bytes) -> char* {
        char* p = ws + off;
        off = (off + bytes + 255) & ~(size_t)255;
        return p;
    };
    _Float16* aw1T   = (_Float16*)take((size_t)64 * 128 * 2);
    _Float16* aw2T   = (_Float16*)take((size_t)16 * 64 * 2);
    _Float16* dw1T   = (_Float16*)take((size_t)128 * DNN_IN * 2);
    _Float16* dw2T   = (_Float16*)take((size_t)64 * 128 * 2);
    _Float16* a_in   = (_Float16*)take((size_t)MROWS * 128 * 2);
    _Float16* h1     = (_Float16*)take((size_t)MROWS * 64 * 2);
    float*    scores = (float*)   take((size_t)MROWS * 4);
    _Float16* xbuf   = (_Float16*)take((size_t)BATCH * DNN_IN * 2);
    _Float16* h1d    = (_Float16*)take((size_t)BATCH * 128 * 2);
    (void)ws_size; (void)in_sizes; (void)n_in; (void)out_size;

    int conv_elems = 64 * 128 + 16 * 64 + 128 * DNN_IN + 64 * 128;
    convert_weights<<<(conv_elems + 255) / 256, 256, 0, stream>>>(
        aw1, aw2, dw1, dw2, aw1T, aw2T, dw1T, dw2T);
    gather_prep<<<BATCH, 64, 0, stream>>>(dense, sparse, history, emb, xbuf, a_in);
    att_layer1<<<MROWS / 16, 128, 0, stream>>>(a_in, aw1T, ab1, h1);
    att_layer23<<<MROWS / 16 / 8, 256, 0, stream>>>(h1, aw2T, ab2, aw3, ab3, scores);
    softmax_pool<<<BATCH, 32, 0, stream>>>(scores, hist_len, a_in, xbuf);
    dnn_layer1<<<BATCH / 16, 256, 0, stream>>>(xbuf, dw1T, db1, h1d);
    dnn_layer23<<<BATCH / 16, 128, 0, stream>>>(h1d, dw2T, db2, dw3, db3, out);
}
